// MultiHeadAttention_4870492914244
// MI455X (gfx1250) — matrix operand order repaired and checked by reading, not compiled, not run on hardware
//
#include <hip/hip_runtime.h>
#include <hip/hip_bf16.h>

// ---------------------------------------------------------------------------
// GQA attention block for MI455X (gfx1250, wave32, WMMA f16 16x16x32).
//   D=2048, H=16, G=4, HD=128, B=2, S=2048
// Pipeline: cast / LDS-tiled transpose -> double-buffered WMMA GEMM with
//           TDM (tensor_load_to_lds + s_wait_tensorcnt) LDS staging ->
//           RoPE relayout -> flash attention (TDM K-tiles, LDS WMMA) ->
//           WMMA GEMM output projection (f32 out).
// ---------------------------------------------------------------------------

typedef __attribute__((ext_vector_type(16))) _Float16 v16h;
typedef __attribute__((ext_vector_type(8)))  _Float16 v8h;
typedef __attribute__((ext_vector_type(8)))  float    v8f;
typedef __attribute__((ext_vector_type(4)))  unsigned int u32x4;
typedef __attribute__((ext_vector_type(4)))  int      i32x4;
typedef __attribute__((ext_vector_type(8)))  int      i32x8;

union V16 { v16h v; struct { v8h lo; v8h hi; } p; };

#define WMMA_F32_F16(a, b, c) \
    __builtin_amdgcn_wmma_f32_16x16x32_f16(false, (a), false, (b), (short)0, (c), false, false)

#ifndef __has_builtin
#define __has_builtin(x) 0
#endif
#if __has_builtin(__builtin_amdgcn_tensor_load_to_lds) && \
    __has_builtin(__builtin_amdgcn_s_wait_tensorcnt)
#define HAVE_TDM 1
#else
#define HAVE_TDM 0
#endif

static constexpr int Dm = 2048;   // d_model
static constexpr int Hh = 16;     // heads
static constexpr int Gg = 4;      // kv groups
static constexpr int HD = 128;    // head dim
static constexpr int Bb = 2;
static constexpr int Ss = 2048;
static constexpr int TK = 32;     // key tile for flash attention

#if HAVE_TDM
// ---------------------------------------------------------------------------
// TDM: async-DMA a 2D f16 tile [tile_h rows x tile_w elems] from global
// (row stride = stride elems) into contiguous LDS at lds_off. D# per ISA
// CDNA5 ISA (group0: count/lds/global/type=2;
// group1: data_size=2B, tensor dims, tile dims, dim0 stride). Tracked by
// TENSORcnt; wave-level (EXEC-independent).
// ---------------------------------------------------------------------------
__device__ __forceinline__ void tdm_load_2d(const _Float16* gsrc,
                                            unsigned lds_off,
                                            unsigned tile_w, unsigned tile_h,
                                            unsigned stride,
                                            unsigned tensor_w, unsigned tensor_h) {
    const unsigned long long ga = (unsigned long long)(size_t)gsrc;
    u32x4 g0;
    g0[0] = 1u;                                            // count=1, user mode
    g0[1] = lds_off;                                       // lds_addr [63:32]
    g0[2] = (unsigned)(ga & 0xFFFFFFFFu);                  // global_addr lo
    g0[3] = (unsigned)((ga >> 32) & 0x1FFFFFFu) | (2u << 30);  // addr hi | type=2
    i32x8 g1;
    g1[0] = (int)(1u << 16);                               // data_size=1 (2 bytes)
    g1[1] = (int)((tensor_w & 0xFFFFu) << 16);             // tensor_dim0[15:0]
    g1[2] = (int)(((tensor_w >> 16) & 0xFFFFu) |           // tensor_dim0[31:16]
                  ((tensor_h & 0xFFFFu) << 16));           // tensor_dim1[15:0]
    g1[3] = (int)(((tensor_h >> 16) & 0xFFFFu) |           // tensor_dim1[31:16]
                  ((tile_w & 0xFFFFu) << 16));             // tile_dim0
    g1[4] = (int)(tile_h & 0xFFFFu);                       // tile_dim1 (tile_dim2=0)
    g1[5] = (int)stride;                                   // tensor_dim0_stride lo32
    g1[6] = 0;                                             // stride hi / dim1_stride lo
    g1[7] = 0;
    const i32x4 z4 = {0, 0, 0, 0};
#if __clang_major__ >= 23
    const i32x8 z8 = {0, 0, 0, 0, 0, 0, 0, 0};
    __builtin_amdgcn_tensor_load_to_lds(g0, g1, z4, z4, z8, 0);
#else
    __builtin_amdgcn_tensor_load_to_lds(g0, g1, z4, z4, 0);
#endif
}
#endif  // HAVE_TDM

// ---------------------------------------------------------------------------
// 1) elementwise cast f32 -> f16
// ---------------------------------------------------------------------------
__global__ void cast_f32_to_f16(const float* __restrict__ src,
                                _Float16* __restrict__ dst, long n) {
    long i = (long)blockIdx.x * blockDim.x + threadIdx.x;
    long stride = (long)gridDim.x * blockDim.x;
    for (; i < n; i += stride) dst[i] = (_Float16)src[i];
}

// ---------------------------------------------------------------------------
// 2) cast + transpose, LDS-tiled (coalesced read and write):
//    W[K,N] (f32, in-major) -> Wt[N,K] (f16, out-major). 32x32 tiles.
// ---------------------------------------------------------------------------
__global__ void __launch_bounds__(256)
cast_transpose(const float* __restrict__ W, _Float16* __restrict__ Wt,
               int K, int N) {
    __shared__ _Float16 tile[32][33];   // +1 pad to dodge bank conflicts
    const int tx = threadIdx.x;         // 0..31
    const int ty = threadIdx.y;         // 0..7
    const int k0 = blockIdx.x * 32;
    const int n0 = blockIdx.y * 32;
    #pragma unroll
    for (int i = 0; i < 4; ++i) {       // read: n contiguous across lanes
        const int k = k0 + ty + i * 8;
        tile[ty + i * 8][tx] = (_Float16)W[(size_t)k * N + (n0 + tx)];
    }
    __syncthreads();
    #pragma unroll
    for (int i = 0; i < 4; ++i) {       // write: k contiguous across lanes
        const int n = n0 + ty + i * 8;
        Wt[(size_t)n * K + (k0 + tx)] = tile[tx][ty + i * 8];
    }
}

// ---------------------------------------------------------------------------
// 3) Register-blocked WMMA GEMM (NT): C[M,N] = A[M,K] * Bt[N,K]^T.
//    Block = 256 thr (8 waves, 4x2), computes a 128x128 C tile.
//    TDM path: double-buffered LDS tiles, one wave issues tensor_load_to_lds
//    for step t+1 while all 8 waves compute step t; s_wait_tensorcnt 2
//    overlaps DMA with WMMA. Fallback: synchronous LDS staging.
//    Frag layouts (16-bit WMMA, wave32, hh = lane>=16):
//      A 16x32: lane m=l&15, K offs {hh*8..+7, hh*8+16..+23}
//      B 32x16: lane n=l&15, K offs {hh*8..+7, hh*8+16..+23}
// ---------------------------------------------------------------------------
template <bool HALF_OUT>
__global__ void __launch_bounds__(256)
gemm_nt_wmma(const _Float16* __restrict__ A, const _Float16* __restrict__ Bt,
             void* __restrict__ C, int M, int N, int K) {
    __shared__ _Float16 sA[2][128][32];    // 16 KB (double buffer)
    __shared__ _Float16 sB[2][128][32];    // 16 KB

    const int tid  = threadIdx.x;
    const int lane = tid & 31;
    const int wave = tid >> 5;
    const int hh   = lane >> 4;
    const int lm   = lane & 15;
    const int wm   = wave >> 1;         // 0..3 : 32-row strip
    const int wn   = wave & 1;          // 0..1 : 64-col strip
    const int bm0  = blockIdx.y * 128;
    const int bn0  = blockIdx.x * 128;

    v8f acc[2][4];
    #pragma unroll
    for (int i = 0; i < 2; ++i)
        #pragma unroll
        for (int n = 0; n < 4; ++n) acc[i][n] = (v8f){};

    const int nt = K / 32;

#if HAVE_TDM
    if (wave == 0) {   // prime the pipeline: tiles for step 0 into buffer 0
        tdm_load_2d(A  + (size_t)bm0 * K, (unsigned)(size_t)&sA[0][0][0],
                    32, 128, (unsigned)K, (unsigned)K, (unsigned)M);
        tdm_load_2d(Bt + (size_t)bn0 * K, (unsigned)(size_t)&sB[0][0][0],
                    32, 128, (unsigned)K, (unsigned)K, (unsigned)N);
    }
#else
    const int srow = tid >> 1;          // 0..127
    const int part = tid & 1;           // 0..1
    const _Float16* gA = A  + (size_t)(bm0 + srow) * K + part * 16;
    const _Float16* gB = Bt + (size_t)(bn0 + srow) * K + part * 16;
#endif

    for (int t = 0; t < nt; ++t) {
        const int buf = t & 1;
#if HAVE_TDM
        if (wave == 0) {
            if (t + 1 < nt) {           // issue DMA for next step, other buffer
                const int kn = (t + 1) * 32;
                tdm_load_2d(A  + (size_t)bm0 * K + kn,
                            (unsigned)(size_t)&sA[buf ^ 1][0][0],
                            32, 128, (unsigned)K, (unsigned)K, (unsigned)M);
                tdm_load_2d(Bt + (size_t)bn0 * K + kn,
                            (unsigned)(size_t)&sB[buf ^ 1][0][0],
                            32, 128, (unsigned)K, (unsigned)K, (unsigned)N);
                __builtin_amdgcn_s_wait_tensorcnt(2);  // step t landed
            } else {
                __builtin_amdgcn_s_wait_tensorcnt(0);
            }
        }
        __syncthreads();                // tiles for step t visible to all waves
#else
        __syncthreads();                // previous compute done with LDS
        {
            const int k = t * 32;
            *reinterpret_cast<v8h*>(&sA[0][srow][part * 16]) =
                *reinterpret_cast<const v8h*>(gA + k);
            *reinterpret_cast<v8h*>(&sA[0][srow][part * 16 + 8]) =
                *reinterpret_cast<const v8h*>(gA + k + 8);
            *reinterpret_cast<v8h*>(&sB[0][srow][part * 16]) =
                *reinterpret_cast<const v8h*>(gB + k);
            *reinterpret_cast<v8h*>(&sB[0][srow][part * 16 + 8]) =
                *reinterpret_cast<const v8h*>(gB + k + 8);
            if (k + 64 < K) {
                __builtin_prefetch(gA + k + 64, 0, 0);
                __builtin_prefetch(gB + k + 64, 0, 0);
            }
        }
        __syncthreads();                // tiles visible
#endif

        V16 af[2], bf[4];
        #pragma unroll
        for (int i = 0; i < 2; ++i) {
            const int r = wm * 32 + i * 16 + lm;
            af[i].p.lo = *reinterpret_cast<const v8h*>(&sA[HAVE_TDM ? buf : 0][r][hh * 8]);
            af[i].p.hi = *reinterpret_cast<const v8h*>(&sA[HAVE_TDM ? buf : 0][r][hh * 8 + 16]);
        }
        #pragma unroll
        for (int n = 0; n < 4; ++n) {
            const int r = wn * 64 + n * 16 + lm;
            bf[n].p.lo = *reinterpret_cast<const v8h*>(&sB[HAVE_TDM ? buf : 0][r][hh * 8]);
            bf[n].p.hi = *reinterpret_cast<const v8h*>(&sB[HAVE_TDM ? buf : 0][r][hh * 8 + 16]);
        }
        #pragma unroll
        for (int i = 0; i < 2; ++i)
            #pragma unroll
            for (int n = 0; n < 4; ++n)
                acc[i][n] = WMMA_F32_F16(af[i].v, bf[n].v, acc[i][n]);

#if HAVE_TDM
        __syncthreads();                // compute done: buffer free for DMA
#endif
    }

    // C layout per tile: lane holds n = lm; VGPR j holds m = j + hh*8
    #pragma unroll
    for (int i = 0; i < 2; ++i) {
        #pragma unroll
        for (int n = 0; n < 4; ++n) {
            const int col = bn0 + wn * 64 + n * 16 + lm;
            #pragma unroll
            for (int j = 0; j < 8; ++j) {
                const int row = bm0 + wm * 32 + i * 16 + j + hh * 8;
                if (HALF_OUT)
                    ((_Float16*)C)[(size_t)row * N + col] = (_Float16)acc[i][n][j];
                else
                    ((float*)C)[(size_t)row * N + col] = acc[i][n][j];
            }
        }
    }
}

// ---------------------------------------------------------------------------
// 4) RoPE + relayout: proj[B,S,nh,HD] -> out[B,nh,S,HD]; rotate_half RoPE.
// ---------------------------------------------------------------------------
__global__ void rope_relayout(const _Float16* __restrict__ proj,
                              _Float16* __restrict__ out, int nh, int do_rope) {
    long i = (long)blockIdx.x * blockDim.x + threadIdx.x;
    const long total = (long)Bb * Ss * nh * 64;
    if (i >= total) return;
    const int d    = (int)(i & 63);
    const int head = (int)((i >> 6) % nh);
    const int s    = (int)((i / (64L * nh)) % Ss);
    const int b    = (int)(i / (64L * nh * Ss));

    const _Float16* src = proj + (((size_t)b * Ss + s) * nh + head) * HD;
    float x1 = (float)src[d];
    float x2 = (float)src[d + 64];
    float o1 = x1, o2 = x2;
    if (do_rope) {
        const float inv_freq = __powf(10000.0f, -((float)(2 * d)) / 128.0f);
        const float ang = (float)s * inv_freq;
        const float c = __cosf(ang), sn = __sinf(ang);
        o1 = x1 * c - x2 * sn;
        o2 = x2 * c + x1 * sn;
    }
    _Float16* dst = out + (((size_t)b * nh + head) * Ss + s) * HD;
    dst[d]      = (_Float16)o1;
    dst[d + 64] = (_Float16)o2;
}

// ---------------------------------------------------------------------------
// 5) Flash attention (causal, GQA). Block = 256 thr (8 waves) handles 128
//    query rows of one (b, h). Key tiles DMA'd into LDS via TDM; V staged
//    transposed by the block. Online softmax across lanes.
//    Output ctx stored as [B,S,H,HD] (== [B*S, D] row-major) in f16.
// ---------------------------------------------------------------------------
__global__ void __launch_bounds__(256)
flash_attn(const _Float16* __restrict__ qh,   // [B,H,S,HD]
           const _Float16* __restrict__ kh,   // [B,G,S,HD]
           const _Float16* __restrict__ vh,   // [B,G,S,HD]
           _Float16* __restrict__ ctx) {      // [B,S,H,HD]
    __shared__ _Float16 sK[TK][HD];           // 8 KB: keys, row-major
    __shared__ _Float16 sVt[HD][TK];          // 8 KB: values, transposed
    __shared__ _Float16 sP[8][16][TK];        // 8 KB: per-wave P tiles

    const int b = blockIdx.z, h = blockIdx.y, qblk = blockIdx.x;
    const int g = h / (Hh / Gg);
    const int tid  = threadIdx.x;
    const int lane = tid & 31;
    const int wave = tid >> 5;
    const int hh   = lane >> 4;
    const int lm   = lane & 15;
    const int q0 = qblk * 128 + wave * 16;
    const float scale = 0.08838834764831845f;  // 1/sqrt(128)

    const _Float16* qbase = qh + ((size_t)b * Hh + h) * Ss * HD;
    const _Float16* kbase = kh + ((size_t)b * Gg + g) * Ss * HD;
    const _Float16* vbase = vh + ((size_t)b * Gg + g) * Ss * HD;

    // Preload Q fragments for the full head dim (4 k-steps of 32).
    V16 qf[4];
    {
        const _Float16* qrow = qbase + (size_t)(q0 + lm) * HD;
        #pragma unroll
        for (int kk = 0; kk < 4; ++kk) {
            qf[kk].p.lo = *reinterpret_cast<const v8h*>(qrow + kk * 32 + hh * 8);
            qf[kk].p.hi = *reinterpret_cast<const v8h*>(qrow + kk * 32 + hh * 8 + 16);
        }
    }

    v8f ctxf[8];
    #pragma unroll
    for (int n = 0; n < 8; ++n) ctxf[n] = (v8f){};
    float Mrow[8], Lrow[8];
    #pragma unroll
    for (int j = 0; j < 8; ++j) { Mrow[j] = -1e30f; Lrow[j] = 0.0f; }

    const int ntiles = (qblk + 1) * (128 / TK);  // causal extent of this block
    const int srow = tid >> 3;                    // 0..31 staging row
    const int seg  = tid & 7;                     // 16-half segment

    for (int t = 0; t < ntiles; ++t) {
        const int kb = t * TK;
        __syncthreads();  // previous tile's LDS reads done
        // --- stage K tile (TDM DMA) and V tile (transposed, by all threads) ---
#if HAVE_TDM
        if (wave == 0)
            tdm_load_2d(kbase + (size_t)kb * HD, (unsigned)(size_t)&sK[0][0],
                        HD, TK, HD, HD, (unsigned)Ss);
#else
        {
            const _Float16* gk = kbase + (size_t)(kb + srow) * HD + seg * 16;
            *reinterpret_cast<v8h*>(&sK[srow][seg * 16])     = *reinterpret_cast<const v8h*>(gk);
            *reinterpret_cast<v8h*>(&sK[srow][seg * 16 + 8]) = *reinterpret_cast<const v8h*>(gk + 8);
        }
#endif
        {
            const _Float16* gv = vbase + (size_t)(kb + srow) * HD + seg * 16;
            V16 vv;
            vv.p.lo = *reinterpret_cast<const v8h*>(gv);
            vv.p.hi = *reinterpret_cast<const v8h*>(gv + 8);
            #pragma unroll
            for (int i = 0; i < 16; ++i)
                sVt[seg * 16 + i][srow] = vv.v[i];
        }
#if HAVE_TDM
        if (wave == 0) __builtin_amdgcn_s_wait_tensorcnt(0);
#endif
        __syncthreads();

        // --- scores: S[16 x 32] = Q(16x128) . K^T, two 16-key subtiles ---
        v8f s0 = {}, s1 = {};
        #pragma unroll
        for (int kk = 0; kk < 4; ++kk) {
            V16 b0, b1;
            b0.p.lo = *reinterpret_cast<const v8h*>(&sK[lm][kk * 32 + hh * 8]);
            b0.p.hi = *reinterpret_cast<const v8h*>(&sK[lm][kk * 32 + hh * 8 + 16]);
            b1.p.lo = *reinterpret_cast<const v8h*>(&sK[16 + lm][kk * 32 + hh * 8]);
            b1.p.hi = *reinterpret_cast<const v8h*>(&sK[16 + lm][kk * 32 + hh * 8 + 16]);
            s0 = WMMA_F32_F16(qf[kk].v, b0.v, s0);
            s1 = WMMA_F32_F16(qf[kk].v, b1.v, s1);
        }

        // --- causal mask + online softmax (N dim lives across 16 lanes) ---
        float fac[8];
        #pragma unroll
        for (int j = 0; j < 8; ++j) {
            const int m = j + hh * 8;
            const int qrow = q0 + m;
            float a0 = s0[j] * scale;
            float a1 = s1[j] * scale;
            if (kb + lm > qrow)      a0 = -1e9f;
            if (kb + 16 + lm > qrow) a1 = -1e9f;
            float rm = fmaxf(a0, a1);
            #pragma unroll
            for (int off = 8; off >= 1; off >>= 1)
                rm = fmaxf(rm, __shfl_xor(rm, off, 32));
            const float mnew = fmaxf(Mrow[j], rm);
            fac[j] = __expf(Mrow[j] - mnew);
            Mrow[j] = mnew;
            const float p0 = __expf(a0 - mnew);
            const float p1 = __expf(a1 - mnew);
            float rs = p0 + p1;
            #pragma unroll
            for (int off = 8; off >= 1; off >>= 1)
                rs += __shfl_xor(rs, off, 32);
            Lrow[j] = Lrow[j] * fac[j] + rs;
            sP[wave][m][lm]      = (_Float16)p0;
            sP[wave][m][16 + lm] = (_Float16)p1;
        }
        #pragma unroll
        for (int n = 0; n < 8; ++n)
            #pragma unroll
            for (int j = 0; j < 8; ++j)
                ctxf[n][j] *= fac[j];
        __syncthreads();  // sP visible; also fences sVt reads vs next stage

        // --- ctx += P(16x32) . V(32x128) over 8 column subtiles ---
        V16 pa;
        pa.p.lo = *reinterpret_cast<const v8h*>(&sP[wave][lm][hh * 8]);
        pa.p.hi = *reinterpret_cast<const v8h*>(&sP[wave][lm][hh * 8 + 16]);
        #pragma unroll
        for (int n = 0; n < 8; ++n) {
            V16 vb;
            vb.p.lo = *reinterpret_cast<const v8h*>(&sVt[n * 16 + lm][hh * 8]);
            vb.p.hi = *reinterpret_cast<const v8h*>(&sVt[n * 16 + lm][hh * 8 + 16]);
            ctxf[n] = WMMA_F32_F16(pa.v, vb.v, ctxf[n]);
        }
    }

    // --- normalize and write ctx[B,S,H,HD] ---
    #pragma unroll
    for (int j = 0; j < 8; ++j) {
        const float inv = 1.0f / Lrow[j];
        const int qrow = q0 + j + hh * 8;
        _Float16* dst = ctx + (((size_t)b * Ss + qrow) * Hh + h) * HD;
        #pragma unroll
        for (int n = 0; n < 8; ++n)
            dst[n * 16 + lm] = (_Float16)(ctxf[n][j] * inv);
    }
}

// ---------------------------------------------------------------------------
// host-side launch
// ---------------------------------------------------------------------------
extern "C" void kernel_launch(void* const* d_in, const int* in_sizes, int n_in,
                              void* d_out, int out_size, void* d_ws, size_t ws_size,
                              hipStream_t stream) {
    const float* x  = (const float*)d_in[0];   // [B,S,D]
    const float* Wq = (const float*)d_in[1];   // [D,D]
    const float* Wk = (const float*)d_in[2];   // [D,G*HD]
    const float* Wv = (const float*)d_in[3];   // [D,G*HD]
    const float* Wo = (const float*)d_in[4];   // [D,D]
    // d_in[5] = causal mask (bool) -- computed analytically in-kernel.
    float* out = (float*)d_out;                // [B,S,D] f32

    const long nX   = (long)Bb * Ss * Dm;      // 8.39M
    const long nWq  = (long)Dm * Dm;
    const long nWkv = (long)Dm * Gg * HD;
    const long nKV  = (long)Bb * Gg * Ss * HD;

    _Float16* p = (_Float16*)d_ws;
    _Float16* xh    = p; p += nX;
    _Float16* Wqt   = p; p += nWq;
    _Float16* Wkt   = p; p += nWkv;
    _Float16* Wvt   = p; p += nWkv;
    _Float16* Wot   = p; p += nWq;
    _Float16* qproj = p; p += nX;
    _Float16* kproj = p; p += nKV;
    _Float16* vproj = p; p += nKV;
    _Float16* qhd   = p; p += nX;
    _Float16* khd   = p; p += nKV;
    _Float16* vhd   = p; p += nKV;
    _Float16* ctxh  = p; p += nX;

    // 1) casts / transposes
    cast_f32_to_f16<<<4096, 256, 0, stream>>>(x, xh, nX);
    {
        dim3 blk(32, 8);
        cast_transpose<<<dim3(Dm / 32, Dm / 32), blk, 0, stream>>>(Wq, Wqt, Dm, Dm);
        cast_transpose<<<dim3(Dm / 32, (Gg * HD) / 32), blk, 0, stream>>>(Wk, Wkt, Dm, Gg * HD);
        cast_transpose<<<dim3(Dm / 32, (Gg * HD) / 32), blk, 0, stream>>>(Wv, Wvt, Dm, Gg * HD);
        cast_transpose<<<dim3(Dm / 32, Dm / 32), blk, 0, stream>>>(Wo, Wot, Dm, Dm);
    }

    // 2) projections: [B*S, D] x [D, N], 128x128 C tiles per block
    const int M = Bb * Ss;  // 4096
    gemm_nt_wmma<true><<<dim3(Dm / 128, M / 128), 256, 0, stream>>>(xh, Wqt, qproj, M, Dm, Dm);
    gemm_nt_wmma<true><<<dim3((Gg * HD) / 128, M / 128), 256, 0, stream>>>(xh, Wkt, kproj, M, Gg * HD, Dm);
    gemm_nt_wmma<true><<<dim3((Gg * HD) / 128, M / 128), 256, 0, stream>>>(xh, Wvt, vproj, M, Gg * HD, Dm);

    // 3) RoPE + relayout to head-major
    {
        const long tq = (long)Bb * Ss * Hh * 64;
        const long tk = (long)Bb * Ss * Gg * 64;
        rope_relayout<<<(int)((tq + 255) / 256), 256, 0, stream>>>(qproj, qhd, Hh, 1);
        rope_relayout<<<(int)((tk + 255) / 256), 256, 0, stream>>>(kproj, khd, Gg, 1);
        rope_relayout<<<(int)((tk + 255) / 256), 256, 0, stream>>>(vproj, vhd, Gg, 0);
    }

    // 4) flash attention -> ctxh [B,S,H,HD]
    flash_attn<<<dim3(Ss / 128, Hh, Bb), 256, 0, stream>>>(qhd, khd, vhd, ctxh);

    // 5) output projection (f32 out): [B*S, D] x [D, D]
    gemm_nt_wmma<false><<<dim3(Dm / 128, M / 128), 256, 0, stream>>>(ctxh, Wot, out, M, Dm, Dm);
}
